// AnomalyTransformer_35966056137299
// MI455X (gfx1250) — matrix operand order repaired and checked by reading, not compiled, not run on hardware
//
#include <hip/hip_runtime.h>

// Problem constants (match reference)
#define B_    8
#define S_    512
#define IN_   32
#define D_    512
#define H_    8
#define L_    6
#define DFF_  2048
#define DK_   64
#define LN_EPS 1e-5f
#define KL_EPS 1e-8f

typedef __bf16 bf16_t;
typedef __attribute__((ext_vector_type(16))) __bf16 v16bf;
typedef __attribute__((ext_vector_type(8)))  __bf16 bf16x8;
typedef __attribute__((ext_vector_type(8)))  float   v8f;
typedef __attribute__((ext_vector_type(4)))  unsigned int u32x4;
typedef __attribute__((ext_vector_type(4)))  int v4i;

#if defined(__has_builtin)
#if __has_builtin(__builtin_amdgcn_global_load_async_to_lds_b128)
#define HAVE_ASYNC_LDS 1
#endif
#endif

// ---------------------------------------------------------------------------
// Fragment layouts per CDNA5 ISA 7.12.2 (wmma_f32_16x16x32_bf16):
//   A 16x32 bf16: lane L -> row M = L&15; half = L>>4 selects K in
//                 {half*8 .. +7} U {16+half*8 .. +7}
//   B 32x16 bf16: lane L -> col N = L&15; same K split as A
//   C 16x16 f32 : lane L -> N = L&15; VGPR r -> M = r + (L>>4)*8
// ---------------------------------------------------------------------------

// ===========================================================================
// LDS-staged, double-buffered, async-pipelined GEMM.
//   C[M,N] = alpha * A[M,K] * B[N,K]^T (+bias) ; A,B bf16 row-major (K contig)
//   Workgroup: 128 threads = 4 waves; block tile 128(M) x 64(N).
//   Wave w owns rows [blockM + w*32, +32) -> MT=2 m-tiles x NT=4 n-tiles.
//   B slab (64 rows x 64 K) staged in LDS, shared by all 4 waves,
//   double-buffered via GLOBAL_LOAD_ASYNC_TO_LDS_B128 (ASYNCcnt) when
//   available, else sync global->register->ds_store.
//   Requires: M%128==0, N%64==0, K%64==0.
// ===========================================================================
__global__ __launch_bounds__(128) void gemm_nt_wmma_lds(
    const bf16_t* __restrict__ A, const bf16_t* __restrict__ B,
    const float* __restrict__ bias,
    float* __restrict__ Cf, bf16_t* __restrict__ Cb,
    int K, int lda, int ldb, int ldcf, int ldcb,
    int nH,
    long long sAb, long long sAh, long long sBb, long long sBh,
    long long sCfb, long long sCfh, long long sCbb, long long sCbh,
    float alpha, int flags)
{
    __shared__ bf16_t Bs[2][64][72];   // 72 = 64 + 8 pad -> conflict-free rows

    const int tileN  = blockIdx.x * 64;
    const int blockM = blockIdx.y * 128;
    const int z  = blockIdx.z;
    const int bb = z / nH;
    const int hh = z % nH;

    const bf16_t* Ap = A + bb * sAb + hh * sAh;
    const bf16_t* Bp = B + bb * sBb + hh * sBh;

    const int tid  = threadIdx.x;
    const int wave = tid >> 5;
    const int lane = tid & 31;
    const int half = lane >> 4;
    const int lr   = lane & 15;
    const int tileM = blockM + wave * 32;          // wave covers 32 rows (MT=2)

    const bf16_t* arow0 = Ap + (long long)(tileM + lr) * lda;
    const bf16_t* arow1 = Ap + (long long)(tileM + 16 + lr) * lda;

    // Cooperative B-slab staging: 64 rows x 64 bf16 = 512 x 16B chunks,
    // 128 threads x 4 chunks.
    auto stageB = [&](int buf, int k0) {
#pragma unroll
        for (int i = 0; i < 4; ++i) {
            const int c   = tid + i * 128;
            const int row = c >> 3;
            const int col = (c & 7) * 8;
            const bf16_t* gp = Bp + (long long)(tileN + row) * ldb + k0 + col;
            bf16_t* lp = &Bs[buf][row][col];
#if HAVE_ASYNC_LDS
            __builtin_amdgcn_global_load_async_to_lds_b128(
                (__attribute__((address_space(1))) v4i*)gp,
                (__attribute__((address_space(3))) v4i*)lp, 0, 0);
#else
            *(u32x4*)lp = *(const u32x4*)gp;
#endif
        }
    };
    auto waitStage = [&]() {
#if HAVE_ASYNC_LDS
#if __has_builtin(__builtin_amdgcn_s_wait_asynccnt)
        __builtin_amdgcn_s_wait_asynccnt(0);
#else
        asm volatile("s_wait_asynccnt 0x0" ::: "memory");
#endif
#endif
        __syncthreads();
    };

    v8f acc[2][4];
#pragma unroll
    for (int m = 0; m < 2; ++m)
#pragma unroll
        for (int t = 0; t < 4; ++t) acc[m][t] = (v8f){0,0,0,0,0,0,0,0};

    const int nSlab = K >> 6;          // K-slabs of 64
    stageB(0, 0);
    waitStage();

    for (int s = 0; s < nSlab; ++s) {
        const int k0 = s << 6;
        if (s + 1 < nSlab) stageB((s + 1) & 1, k0 + 64);   // prefetch next slab

        // Prefetch all A fragments for this slab (2 k-steps x 2 m-tiles)
        bf16x8 aflo[2][2], afhi[2][2];
#pragma unroll
        for (int ks = 0; ks < 2; ++ks) {
            aflo[0][ks] = *(const bf16x8*)(arow0 + k0 + ks * 32 + half * 8);
            afhi[0][ks] = *(const bf16x8*)(arow0 + k0 + ks * 32 + 16 + half * 8);
            aflo[1][ks] = *(const bf16x8*)(arow1 + k0 + ks * 32 + half * 8);
            afhi[1][ks] = *(const bf16x8*)(arow1 + k0 + ks * 32 + 16 + half * 8);
        }

        const bf16_t* bbuf = &Bs[s & 1][0][0];
#pragma unroll
        for (int ks = 0; ks < 2; ++ks) {
            v16bf a[2];
#pragma unroll
            for (int m = 0; m < 2; ++m)
#pragma unroll
                for (int j = 0; j < 8; ++j) {
                    a[m][j]     = aflo[m][ks][j];
                    a[m][8 + j] = afhi[m][ks][j];
                }
#pragma unroll
            for (int t = 0; t < 4; ++t) {
                const bf16_t* bp = bbuf + (t * 16 + lr) * 72 + ks * 32 + half * 8;
                bf16x8 blo = *(const bf16x8*)bp;
                bf16x8 bhi = *(const bf16x8*)(bp + 16);
                v16bf b;
#pragma unroll
                for (int j = 0; j < 8; ++j) { b[j] = blo[j]; b[8 + j] = bhi[j]; }
#pragma unroll
                for (int m = 0; m < 2; ++m)
                    acc[m][t] = __builtin_amdgcn_wmma_f32_16x16x32_bf16(
                        false, a[m], false, b, (short)0, acc[m][t], false, false);
            }
        }
        waitStage();   // drain async copies for next slab; release current buf
    }

    float*  cfp = Cf ? Cf + bb * sCfb + hh * sCfh : nullptr;
    bf16_t* cbp = Cb ? Cb + bb * sCbb + hh * sCbh : nullptr;
#pragma unroll
    for (int m = 0; m < 2; ++m) {
#pragma unroll
        for (int t = 0; t < 4; ++t) {
            const int n_out = tileN + t * 16 + lr;
            const float bv = bias ? bias[n_out] : 0.0f;
#pragma unroll
            for (int r = 0; r < 8; ++r) {
                const int m_out = tileM + m * 16 + half * 8 + r;
                float v = acc[m][t][r] * alpha + bv;
                if (flags & 1) v = v > 0.0f ? v : 0.0f;          // relu
                if (cfp) cfp[(long long)m_out * ldcf + n_out] = v;
                if (cbp) {
                    if (flags & 2) cbp[(long long)n_out * ldcb + m_out] = (bf16_t)v;
                    else           cbp[(long long)m_out * ldcb + n_out] = (bf16_t)v;
                }
            }
        }
    }
}

// ===========================================================================
// Direct-load fallback GEMM (used for N=32 reconstruction head).
// ===========================================================================
template<int NT>
__global__ __launch_bounds__(32) void gemm_nt_wmma(
    const bf16_t* __restrict__ A, const bf16_t* __restrict__ B,
    const float* __restrict__ bias,
    float* __restrict__ Cf, bf16_t* __restrict__ Cb,
    int K, int lda, int ldb, int ldcf, int ldcb,
    int nH,
    long long sAb, long long sAh, long long sBb, long long sBh,
    long long sCfb, long long sCfh, long long sCbb, long long sCbh,
    float alpha, int flags)
{
    const int tileN = blockIdx.x * (16 * NT);
    const int tileM = blockIdx.y * 16;
    const int z  = blockIdx.z;
    const int bb = z / nH;
    const int hh = z % nH;

    const bf16_t* Ap = A + bb * sAb + hh * sAh;
    const bf16_t* Bp = B + bb * sBb + hh * sBh;

    const int lane = threadIdx.x;
    const int half = lane >> 4;
    const int lr   = lane & 15;

    const bf16_t* arow = Ap + (long long)(tileM + lr) * lda;
    const bf16_t* brow[NT];
#pragma unroll
    for (int t = 0; t < NT; ++t)
        brow[t] = Bp + (long long)(tileN + t * 16 + lr) * ldb + half * 8;

    v8f acc[NT];
#pragma unroll
    for (int t = 0; t < NT; ++t) acc[t] = (v8f){0,0,0,0,0,0,0,0};

    for (int k0 = 0; k0 < K; k0 += 32) {
        bf16x8 alo = *(const bf16x8*)(arow + k0 + half * 8);
        bf16x8 ahi = *(const bf16x8*)(arow + k0 + 16 + half * 8);
        v16bf a;
#pragma unroll
        for (int j = 0; j < 8; ++j) { a[j] = alo[j]; a[8 + j] = ahi[j]; }
#pragma unroll
        for (int t = 0; t < NT; ++t) {
            bf16x8 blo = *(const bf16x8*)(brow[t] + k0);
            bf16x8 bhi = *(const bf16x8*)(brow[t] + k0 + 16);
            v16bf b;
#pragma unroll
            for (int j = 0; j < 8; ++j) { b[j] = blo[j]; b[8 + j] = bhi[j]; }
            acc[t] = __builtin_amdgcn_wmma_f32_16x16x32_bf16(
                false, a, false, b, (short)0, acc[t], false, false);
        }
    }

    float*  cfp = Cf ? Cf + bb * sCfb + hh * sCfh : nullptr;
    bf16_t* cbp = Cb ? Cb + bb * sCbb + hh * sCbh : nullptr;
#pragma unroll
    for (int t = 0; t < NT; ++t) {
        const int n_out = tileN + t * 16 + lr;
        const float bv = bias ? bias[n_out] : 0.0f;
#pragma unroll
        for (int r = 0; r < 8; ++r) {
            const int m_out = tileM + half * 8 + r;
            float v = acc[t][r] * alpha + bv;
            if (flags & 1) v = v > 0.0f ? v : 0.0f;
            if (cfp) cfp[(long long)m_out * ldcf + n_out] = v;
            if (cbp) {
                if (flags & 2) cbp[(long long)n_out * ldcb + m_out] = (bf16_t)v;
                else           cbp[(long long)m_out * ldcb + n_out] = (bf16_t)v;
            }
        }
    }
}

// W[K,N] fp32  ->  Wt[N,K] bf16 (rows contiguous in K)
__global__ __launch_bounds__(256) void transpose_to_bf16(
    const float* __restrict__ W, bf16_t* __restrict__ Wt, int K, int N)
{
    long long idx = (long long)blockIdx.x * 256 + threadIdx.x;
    if (idx >= (long long)K * N) return;
    int n = (int)(idx / K);
    int k = (int)(idx % K);
    Wt[idx] = (bf16_t)W[(long long)k * N + n];
}

// h = x @ W_in + b_in + pe  (K = 32, cheap) ; also emits bf16 copy
__global__ __launch_bounds__(256) void embed_kernel(
    const float* __restrict__ x, const float* __restrict__ W_in,
    const float* __restrict__ b_in, const float* __restrict__ pe,
    float* __restrict__ h, bf16_t* __restrict__ hb)
{
    long long idx = (long long)blockIdx.x * 256 + threadIdx.x;
    if (idx >= (long long)B_ * S_ * D_) return;
    int d   = (int)(idx % D_);
    int row = (int)(idx / D_);
    int s   = row % S_;
    const float* xr = x + (long long)row * IN_;
    float acc = b_in[d] + pe[(long long)s * D_ + d];
    for (int k = 0; k < IN_; ++k) acc += xr[k] * W_in[(long long)k * D_ + d];
    h[idx]  = acc;
    hb[idx] = (bf16_t)acc;
}

// row softmax over S_ elems; writes fp32 probs in place + bf16 copy
__global__ __launch_bounds__(256) void softmax_rows(
    float* __restrict__ scores, bf16_t* __restrict__ attnb)
{
    __shared__ float red[256];
    const int row = blockIdx.x;                       // b*H*S + h*S + s
    float*  p  = scores + (long long)row * S_;
    bf16_t* pb = attnb  + (long long)row * S_;
    const int t = threadIdx.x;

    float m = -3.0e38f;
    for (int i = t; i < S_; i += 256) m = fmaxf(m, p[i]);
    red[t] = m; __syncthreads();
    for (int o = 128; o > 0; o >>= 1) { if (t < o) red[t] = fmaxf(red[t], red[t + o]); __syncthreads(); }
    m = red[0]; __syncthreads();

    float sum = 0.0f;
    for (int i = t; i < S_; i += 256) { float e = __expf(p[i] - m); p[i] = e; sum += e; }
    red[t] = sum; __syncthreads();
    for (int o = 128; o > 0; o >>= 1) { if (t < o) red[t] += red[t + o]; __syncthreads(); }
    const float inv = 1.0f / red[0];

    for (int i = t; i < S_; i += 256) { float v = p[i] * inv; p[i] = v; pb[i] = (bf16_t)v; }
}

// total_assoc[b,s] += sum_t (avg_h attn + eps) * log((avg_h attn + eps)/prior)
__global__ __launch_bounds__(256) void avg_kl_kernel(
    const float* __restrict__ probs, float* __restrict__ assoc)
{
    __shared__ float red[256];
    const int row = blockIdx.x;                       // b*S + s
    const int b = row / S_, s = row % S_;
    const float* base = probs + (long long)b * H_ * S_ * S_ + (long long)s * S_;
    const float prior = 1.0f / (float)S_ + KL_EPS;
    float acc = 0.0f;
    for (int t = threadIdx.x; t < S_; t += 256) {
        float av = 0.0f;
        for (int hh = 0; hh < H_; ++hh) av += base[(long long)hh * S_ * S_ + t];
        av = av * (1.0f / H_) + KL_EPS;
        acc += av * __logf(av / prior);
    }
    red[threadIdx.x] = acc; __syncthreads();
    for (int o = 128; o > 0; o >>= 1) { if (threadIdx.x < o) red[threadIdx.x] += red[threadIdx.x + o]; __syncthreads(); }
    if (threadIdx.x == 0) assoc[row] += red[0];
}

// h = LayerNorm(h + y) * g + b ; also emits bf16 copy
__global__ __launch_bounds__(256) void ln_residual(
    float* __restrict__ h, const float* __restrict__ y,
    const float* __restrict__ g, const float* __restrict__ be,
    bf16_t* __restrict__ hb)
{
    __shared__ float red[256];
    const int row = blockIdx.x;                       // b*S + s
    float* hp = h + (long long)row * D_;
    const float* yp = y + (long long)row * D_;
    const int t = threadIdx.x;
    float v0 = hp[t] + yp[t];
    float v1 = hp[t + 256] + yp[t + 256];

    red[t] = v0 + v1; __syncthreads();
    for (int o = 128; o > 0; o >>= 1) { if (t < o) red[t] += red[t + o]; __syncthreads(); }
    const float mu = red[0] * (1.0f / D_); __syncthreads();

    const float d0 = v0 - mu, d1 = v1 - mu;
    red[t] = d0 * d0 + d1 * d1; __syncthreads();
    for (int o = 128; o > 0; o >>= 1) { if (t < o) red[t] += red[t + o]; __syncthreads(); }
    const float rs = rsqrtf(red[0] * (1.0f / D_) + LN_EPS);

    const float o0 = d0 * rs * g[t] + be[t];
    const float o1 = d1 * rs * g[t + 256] + be[t + 256];
    hp[t] = o0; hp[t + 256] = o1;
    hb[(long long)row * D_ + t]       = (bf16_t)o0;
    hb[(long long)row * D_ + t + 256] = (bf16_t)o1;
}

// anomaly = sigmoid(s1 @ Ws2 + bs2)
__global__ __launch_bounds__(256) void anomaly_head(
    const bf16_t* __restrict__ s1, const float* __restrict__ Ws2,
    const float* __restrict__ bs2, float* __restrict__ out)
{
    int row = blockIdx.x * 256 + threadIdx.x;
    if (row >= B_ * S_) return;
    const bf16_t* r = s1 + (long long)row * (D_ / 2);
    float acc = bs2[0];
    for (int j = 0; j < D_ / 2; ++j) acc += (float)r[j] * Ws2[j];
    out[row] = 1.0f / (1.0f + __expf(-acc));
}

__global__ __launch_bounds__(256) void finalize_assoc(
    const float* __restrict__ assoc, float* __restrict__ out)
{
    int i = blockIdx.x * 256 + threadIdx.x;
    if (i < B_ * S_) out[i] = assoc[i] * (1.0f / L_);
}

__global__ __launch_bounds__(256) void zero_f32(float* p, int n)
{
    int i = blockIdx.x * 256 + threadIdx.x;
    if (i < n) p[i] = 0.0f;
}

// ---------------------------------------------------------------------------
// Host-side GEMM dispatch: LDS-pipelined kernel when tiles permit, else direct
// ---------------------------------------------------------------------------
static void run_gemm(hipStream_t st, const bf16_t* A, const bf16_t* Bm, const float* bias,
                     float* Cf, bf16_t* Cb, int M, int N, int K,
                     int lda, int ldb, int ldcf, int ldcb,
                     int nB, int nH,
                     long long sAb, long long sAh, long long sBb, long long sBh,
                     long long sCfb, long long sCfh, long long sCbb, long long sCbh,
                     float alpha, int flags)
{
    if ((M % 128 == 0) && (N % 64 == 0) && (K % 64 == 0)) {
        dim3 g(N / 64, M / 128, nB * nH);
        gemm_nt_wmma_lds<<<g, 128, 0, st>>>(A, Bm, bias, Cf, Cb, K, lda, ldb, ldcf, ldcb,
            nH, sAb, sAh, sBb, sBh, sCfb, sCfh, sCbb, sCbh, alpha, flags);
    } else if (N % 32 == 0) {
        dim3 g(N / 32, M / 16, nB * nH);
        gemm_nt_wmma<2><<<g, 32, 0, st>>>(A, Bm, bias, Cf, Cb, K, lda, ldb, ldcf, ldcb,
            nH, sAb, sAh, sBb, sBh, sCfb, sCfh, sCbb, sCbh, alpha, flags);
    } else {
        dim3 g(N / 16, M / 16, nB * nH);
        gemm_nt_wmma<1><<<g, 32, 0, st>>>(A, Bm, bias, Cf, Cb, K, lda, ldb, ldcf, ldcb,
            nH, sAb, sAh, sBb, sBh, sCfb, sCfh, sCbb, sCbh, alpha, flags);
    }
}

extern "C" void kernel_launch(void* const* d_in, const int* in_sizes, int n_in,
                              void* d_out, int out_size, void* d_ws, size_t ws_size,
                              hipStream_t stream)
{
    (void)in_sizes; (void)n_in; (void)out_size; (void)ws_size;

    const float* x     = (const float*)d_in[0];
    const float* W_in  = (const float*)d_in[1];
    const float* b_in  = (const float*)d_in[2];
    const float* pe    = (const float*)d_in[3];
    const float* Wq    = (const float*)d_in[4];
    const float* Wk    = (const float*)d_in[5];
    const float* Wv    = (const float*)d_in[6];
    const float* Wo    = (const float*)d_in[7];
    const float* bo    = (const float*)d_in[8];
    const float* W1    = (const float*)d_in[9];
    const float* b1    = (const float*)d_in[10];
    const float* W2    = (const float*)d_in[11];
    const float* b2    = (const float*)d_in[12];
    const float* ln1_g = (const float*)d_in[13];
    const float* ln1_b = (const float*)d_in[14];
    const float* ln2_g = (const float*)d_in[15];
    const float* ln2_b = (const float*)d_in[16];
    const float* W_out = (const float*)d_in[17];
    const float* b_out = (const float*)d_in[18];
    const float* Ws1   = (const float*)d_in[19];
    const float* bs1   = (const float*)d_in[20];
    const float* Ws2   = (const float*)d_in[21];
    const float* bs2   = (const float*)d_in[22];

    const long long M  = (long long)B_ * S_;          // 4096
    const long long MD = M * D_;                       // 2,097,152

    // Workspace carve-out (256B aligned slabs)
    char* ws = (char*)d_ws;
    size_t off = 0;
    auto alloc = [&](size_t bytes) -> void* {
        off = (off + 255) & ~(size_t)255;
        void* p = ws + off;
        off += bytes;
        return p;
    };

    float*  h      = (float*) alloc(MD * 4);
    bf16_t* hb     = (bf16_t*)alloc(MD * 2);
    bf16_t* qb     = (bf16_t*)alloc(MD * 2);
    bf16_t* kb     = (bf16_t*)alloc(MD * 2);
    bf16_t* vtb    = (bf16_t*)alloc(MD * 2);           // V transposed: [D, B*S]
    bf16_t* ctxb   = (bf16_t*)alloc(MD * 2);
    bf16_t* ff1b   = (bf16_t*)alloc((size_t)M * DFF_ * 2);
    float*  tmp    = (float*) alloc(MD * 4);           // attn_out / ffn_out
    float*  scores = (float*) alloc((size_t)B_ * H_ * S_ * S_ * 4);
    bf16_t* attnb  = (bf16_t*)alloc((size_t)B_ * H_ * S_ * S_ * 2);
    float*  assoc  = (float*) alloc((size_t)M * 4);
    bf16_t* WqT    = (bf16_t*)alloc((size_t)D_ * D_ * 2);
    bf16_t* WkT    = (bf16_t*)alloc((size_t)D_ * D_ * 2);
    bf16_t* WvT    = (bf16_t*)alloc((size_t)D_ * D_ * 2);
    bf16_t* WoT    = (bf16_t*)alloc((size_t)D_ * D_ * 2);
    bf16_t* W1T    = (bf16_t*)alloc((size_t)D_ * DFF_ * 2);
    bf16_t* W2T    = (bf16_t*)alloc((size_t)DFF_ * D_ * 2);
    bf16_t* WoutT  = (bf16_t*)alloc((size_t)D_ * IN_ * 2);
    bf16_t* Ws1T   = (bf16_t*)alloc((size_t)D_ * (D_ / 2) * 2);
    bf16_t* s1b    = (bf16_t*)alloc((size_t)M * (D_ / 2) * 2);

    float* out_recon = (float*)d_out;                  // [B,S,IN]
    float* out_anom  = out_recon + (size_t)B_ * S_ * IN_;
    float* out_assoc = out_anom + (size_t)B_ * S_;

    const float scale = 0.125f;                        // 1/sqrt(DK=64)
    const long long sQh = 64, sQb = (long long)S_ * D_;
    const long long sSc = (long long)S_ * S_;

    // --- prologue ---
    zero_f32<<<dim3((B_ * S_ + 255) / 256), 256, 0, stream>>>(assoc, B_ * S_);
    embed_kernel<<<dim3((unsigned)((MD + 255) / 256)), 256, 0, stream>>>(x, W_in, b_in, pe, h, hb);

    for (int l = 0; l < L_; ++l) {
        const long long wdd = (long long)l * D_ * D_;
        unsigned gdd  = (unsigned)(((size_t)D_ * D_   + 255) / 256);
        unsigned gdff = (unsigned)(((size_t)D_ * DFF_ + 255) / 256);
        transpose_to_bf16<<<gdd,  256, 0, stream>>>(Wq + wdd, WqT, D_, D_);
        transpose_to_bf16<<<gdd,  256, 0, stream>>>(Wk + wdd, WkT, D_, D_);
        transpose_to_bf16<<<gdd,  256, 0, stream>>>(Wv + wdd, WvT, D_, D_);
        transpose_to_bf16<<<gdd,  256, 0, stream>>>(Wo + wdd, WoT, D_, D_);
        transpose_to_bf16<<<gdff, 256, 0, stream>>>(W1 + (long long)l * D_ * DFF_, W1T, D_,   DFF_);
        transpose_to_bf16<<<gdff, 256, 0, stream>>>(W2 + (long long)l * DFF_ * D_, W2T, DFF_, D_);

        // Q, K (bf16, [B*S, D]); V stored transposed ([D, B*S]) for ctx GEMM
        run_gemm(stream, hb, WqT, nullptr, nullptr, qb,  (int)M, D_, D_, D_, D_, 0, D_,
                 1, 1, 0,0,0,0, 0,0,0,0, 1.0f, 0);
        run_gemm(stream, hb, WkT, nullptr, nullptr, kb,  (int)M, D_, D_, D_, D_, 0, D_,
                 1, 1, 0,0,0,0, 0,0,0,0, 1.0f, 0);
        run_gemm(stream, hb, WvT, nullptr, nullptr, vtb, (int)M, D_, D_, D_, D_, 0, (int)M,
                 1, 1, 0,0,0,0, 0,0,0,0, 1.0f, 2 /*transposed store*/);

        // scores[b,h,s,t] = scale * Q_h K_h^T   (batched over B*H)
        run_gemm(stream, qb, kb, nullptr, scores, nullptr, S_, S_, DK_,
                 D_, D_, S_, 0, B_, H_,
                 sQb, sQh, sQb, sQh, (long long)H_ * sSc, sSc, 0, 0, scale, 0);

        softmax_rows<<<dim3(B_ * H_ * S_), 256, 0, stream>>>(scores, attnb);
        avg_kl_kernel<<<dim3(B_ * S_), 256, 0, stream>>>(scores, assoc);

        // ctx[b,s,h*64+k] = attn @ V  (A = attn bf16, B = Vt rows contiguous in t)
        run_gemm(stream, attnb, vtb, nullptr, nullptr, ctxb, S_, DK_, S_,
                 S_, (int)M, 0, D_, B_, H_,
                 (long long)H_ * sSc, sSc, (long long)S_, (long long)DK_ * M,
                 0, 0, sQb, sQh, 1.0f, 0);

        // attn_out = ctx @ Wo + bo  -> tmp (f32)
        run_gemm(stream, ctxb, WoT, bo + (long long)l * D_, tmp, nullptr,
                 (int)M, D_, D_, D_, D_, D_, 0, 1, 1, 0,0,0,0, 0,0,0,0, 1.0f, 0);

        ln_residual<<<dim3((unsigned)M), 256, 0, stream>>>(
            h, tmp, ln1_g + (long long)l * D_, ln1_b + (long long)l * D_, hb);

        // FFN
        run_gemm(stream, hb, W1T, b1 + (long long)l * DFF_, nullptr, ff1b,
                 (int)M, DFF_, D_, D_, D_, 0, DFF_, 1, 1, 0,0,0,0, 0,0,0,0, 1.0f, 1 /*relu*/);
        run_gemm(stream, ff1b, W2T, b2 + (long long)l * D_, tmp, nullptr,
                 (int)M, D_, DFF_, DFF_, DFF_, D_, 0, 1, 1, 0,0,0,0, 0,0,0,0, 1.0f, 0);

        ln_residual<<<dim3((unsigned)M), 256, 0, stream>>>(
            h, tmp, ln2_g + (long long)l * D_, ln2_b + (long long)l * D_, hb);
    }

    // --- heads ---
    unsigned gout = (unsigned)(((size_t)D_ * IN_ + 255) / 256);
    unsigned gs1  = (unsigned)(((size_t)D_ * (D_ / 2) + 255) / 256);
    transpose_to_bf16<<<gout, 256, 0, stream>>>(W_out, WoutT, D_, IN_);
    transpose_to_bf16<<<gs1,  256, 0, stream>>>(Ws1,   Ws1T,  D_, D_ / 2);

    // reconstruction -> d_out (f32, ldc = IN)
    run_gemm(stream, hb, WoutT, b_out, out_recon, nullptr,
             (int)M, IN_, D_, D_, D_, IN_, 0, 1, 1, 0,0,0,0, 0,0,0,0, 1.0f, 0);

    // s1 = relu(h @ Ws1 + bs1) (bf16), then sigmoid head + assoc finalize
    run_gemm(stream, hb, Ws1T, bs1, nullptr, s1b,
             (int)M, D_ / 2, D_, D_, D_, 0, D_ / 2, 1, 1, 0,0,0,0, 0,0,0,0, 1.0f, 1 /*relu*/);

    anomaly_head<<<dim3((B_ * S_ + 255) / 256), 256, 0, stream>>>(s1b, Ws2, bs2, out_anom);
    finalize_assoc<<<dim3((B_ * S_ + 255) / 256), 256, 0, stream>>>(assoc, out_assoc);
}
